// MetaLSTMDetector_36412732735925
// MI455X (gfx1250) — matrix operand order repaired and checked by reading, not compiled, not run on hardware
//
#include <hip/hip_runtime.h>

// ---------------- problem constants ----------------
#define H        256
#define T_LEN    4096
#define GROUPS   4      // batch groups (16 batch rows each)
#define WPG      16     // workgroups per group (each owns 16 hidden units)
#define BG       16     // batch rows per group  (WMMA M dim)
#define NH       16     // hidden units per workgroup
#define NTHREADS 128    // 4 wave32 -> one wave per gate type (i,f,g,o)
#define PAD_VAL  -100.0f

typedef __attribute__((ext_vector_type(16))) _Float16 v16h;
typedef __attribute__((ext_vector_type(8)))  _Float16 v8h;
typedef __attribute__((ext_vector_type(8)))  float    v8f;
typedef int v4i __attribute__((ext_vector_type(4)));

__device__ __forceinline__ float fsigmoid(float x) {
    return 1.0f / (1.0f + __expf(-x));
}
__device__ __forceinline__ float ftanh(float x) {
    return 2.0f / (1.0f + __expf(-2.0f * x)) - 1.0f;
}

// K=256 GEMM accumulate: D += A(16x256 LDS, row-major) x B(16x256 LDS rows = W slice)^T
// using 8x v_wmma_f32_16x16x32_f16.
__device__ __forceinline__ v8f wmma_k256(const _Float16* __restrict__ A,
                                         const _Float16* __restrict__ Bw,
                                         v8f acc, int lane) {
    const int mn = lane & 15;
    const int kh = lane >> 4;
    const _Float16* arow = A  + mn * H + kh * 16;
    const _Float16* brow = Bw + mn * H + kh * 16;
#pragma unroll
    for (int k0 = 0; k0 < H; k0 += 32) {
        v8h alo = *(const v8h*)(arow + k0);
        v8h ahi = *(const v8h*)(arow + k0 + 8);
        v16h a = __builtin_shufflevector(alo, ahi,
                 0,1,2,3,4,5,6,7,8,9,10,11,12,13,14,15);
        v16h b = *(const v16h*)(brow + k0);
        acc = __builtin_amdgcn_wmma_f32_16x16x32_f16(
                  false, a, false, b, (short)0, acc, false, false);
    }
    return acc;
}

// Gather a BG x H f16 block (8 KB) from L2 into LDS.
// Preferred path: CDNA5 async direct-to-LDS loads (ASYNCcnt tracked, no VGPR
// round trip). Fallback: plain b128 load + ds_store copy.
__device__ __forceinline__ void gather_to_lds(_Float16* dstLds,
                                              const _Float16* srcGlb,
                                              int tid) {
#if defined(__gfx1250__) && __has_builtin(__builtin_amdgcn_global_load_async_to_lds_b128)
    // builtin takes generic (flat) v4i* for both source and LDS destination
    v4i* src = (v4i*)srcGlb;
    v4i* dst = (v4i*)dstLds;
#pragma unroll
    for (int idx = tid; idx < BG * H * 2 / 16; idx += NTHREADS)
        __builtin_amdgcn_global_load_async_to_lds_b128(src + idx, dst + idx, 0, 0);
#if __has_builtin(__builtin_amdgcn_s_wait_asynccnt)
    __builtin_amdgcn_s_wait_asynccnt(0);
#else
    asm volatile("s_wait_asynccnt 0x0" ::: "memory");
#endif
#else
    const uint4* src = (const uint4*)srcGlb;
    uint4* dst = (uint4*)dstLds;
    for (int idx = tid; idx < BG * H * 2 / 16; idx += NTHREADS)
        dst[idx] = src[idx];
#endif
}

// generation-counting barrier across the 16 workgroups of one batch group.
__device__ __forceinline__ void group_barrier(unsigned* cnt, unsigned* gen,
                                              unsigned target) {
    __syncthreads();
    if (threadIdx.x == 0) {
        __threadfence();
        unsigned prev = atomicAdd(cnt, 1u);
        if (prev == WPG - 1) {
            atomicExch(cnt, 0u);
            __threadfence();
            atomicAdd(gen, 1u);          // release
        } else {
            while (atomicAdd(gen, 0u) < target) {
                __builtin_amdgcn_s_sleep(1);
            }
        }
        __threadfence();
    }
    __syncthreads();
}

__global__ void __launch_bounds__(NTHREADS)
meta_lstm_kernel(const float* __restrict__ y,
                 const float* __restrict__ W_ih0, const float* __restrict__ W_hh0,
                 const float* __restrict__ b_ih0, const float* __restrict__ b_hh0,
                 const float* __restrict__ W_ih1, const float* __restrict__ W_hh1,
                 const float* __restrict__ b_ih1, const float* __restrict__ b_hh1,
                 const float* __restrict__ W_out, const float* __restrict__ b_out,
                 float* __restrict__ out, void* __restrict__ ws) {
    // LDS-resident weight slices (64 gate rows x 256, f16) + activations.
    __shared__ __align__(32) _Float16 Whh0L[64 * H];
    __shared__ __align__(32) _Float16 Wih1L[64 * H];
    __shared__ __align__(32) _Float16 Whh1L[64 * H];
    __shared__ __align__(32) _Float16 h0full[BG * H];
    __shared__ __align__(32) _Float16 h1full[BG * H];
    __shared__ float gbuf[4][16][16];    // per-gate-type WMMA results
    __shared__ float cbuf0[16][16];
    __shared__ float cbuf1[16][16];
    __shared__ float xbuf[16][4];
    __shared__ float Wih0L[64 * 4];
    __shared__ float bias0L[64];
    __shared__ float bias1L[64];
    __shared__ float WoutL[2 * H];
    __shared__ float boutL[2];

    const int tid   = threadIdx.x;
    const int g     = blockIdx.x / WPG;   // batch group
    const int s     = blockIdx.x % WPG;   // hidden slice within group
    const int hbase = s * NH;
    const int lane  = tid & 31;
    const int wv    = tid >> 5;           // gate type 0..3 (i,f,g,o)

    unsigned* bar   = (unsigned*)ws;
    _Float16* h0buf = (_Float16*)((char*)ws + 256);
    _Float16* h1buf = h0buf + GROUPS * BG * H;
    _Float16* myH0  = h0buf + g * BG * H;
    _Float16* myH1  = h1buf + g * BG * H;
    unsigned* cnt   = bar + g * 2;
    unsigned* gen   = bar + g * 2 + 1;

    // ---- one-time weight preload: f32 -> f16 into LDS ----
    for (int idx = tid; idx < 64 * H; idx += NTHREADS) {
        int row = idx >> 8, k = idx & (H - 1);
        int type = row >> 4, r = row & 15;
        int grow = type * H + hbase + r;          // global gate row
        Whh0L[idx] = (_Float16)W_hh0[grow * H + k];
        Wih1L[idx] = (_Float16)W_ih1[grow * H + k];
        Whh1L[idx] = (_Float16)W_hh1[grow * H + k];
    }
    if (tid < 64) {
        int type = tid >> 4, r = tid & 15;
        int grow = type * H + hbase + r;
        bias0L[tid] = b_ih0[grow] + b_hh0[grow];
        bias1L[tid] = b_ih1[grow] + b_hh1[grow];
#pragma unroll
        for (int j = 0; j < 4; ++j) Wih0L[tid * 4 + j] = W_ih0[grow * 4 + j];
    }
    for (int idx = tid; idx < 2 * H; idx += NTHREADS) WoutL[idx] = W_out[idx];
    if (tid < 2) boutL[tid] = b_out[tid];
    for (int idx = tid; idx < BG * H; idx += NTHREADS) {
        h0full[idx] = (_Float16)0.0f;
        h1full[idx] = (_Float16)0.0f;
    }
    for (int idx = tid; idx < 256; idx += NTHREADS) {
        cbuf0[idx >> 4][idx & 15] = 0.0f;
        cbuf1[idx >> 4][idx & 15] = 0.0f;
    }
    __syncthreads();

    unsigned bgen = 0;

    for (int t = 0; t < T_LEN; ++t) {
        // causal input window x_t = [y_{t-3}..y_t] with front pad
        if (tid < 64) {
            int m = tid >> 2, j = tid & 3;
            int tt = t - 3 + j;
            xbuf[m][j] = (tt >= 0) ? y[(g * BG + m) * T_LEN + tt] : PAD_VAL;
        }
        __syncthreads();

        // ======== layer 0: gates slice = h0(t-1) @ W_hh0^T ========
        {
            v8f acc = {};
            acc = wmma_k256(h0full, Whh0L + wv * 16 * H, acc, lane);
            const int n = lane & 15, kh = lane >> 4;
#pragma unroll
            for (int j = 0; j < 8; ++j) gbuf[wv][j + 8 * kh][n] = acc[j];
        }
        __syncthreads();

        // elementwise layer0: fold K=4 input GEMM + biases, update c0, h0
        for (int e = tid; e < 256; e += NTHREADS) {
            int m = e >> 4, n = e & 15;
            float gv[4];
#pragma unroll
            for (int q = 0; q < 4; ++q) {
                float gs = gbuf[q][m][n] + bias0L[q * 16 + n];
#pragma unroll
                for (int j = 0; j < 4; ++j)
                    gs += Wih0L[(q * 16 + n) * 4 + j] * xbuf[m][j];
                gv[q] = gs;
            }
            float cn = fsigmoid(gv[1]) * cbuf0[m][n]
                     + fsigmoid(gv[0]) * ftanh(gv[2]);
            cbuf0[m][n] = cn;
            float hn = fsigmoid(gv[3]) * ftanh(cn);
            myH0[m * H + hbase + n] = (_Float16)hn;   // publish slice
        }
        group_barrier(cnt, gen, ++bgen);

        // gather full h0(t) for layer 1 (async direct-to-LDS)
        gather_to_lds(h0full, myH0, tid);
        __syncthreads();

        // ======== layer 1: gates = h0(t) @ W_ih1^T + h1(t-1) @ W_hh1^T ========
        {
            v8f acc = {};
            acc = wmma_k256(h0full, Wih1L + wv * 16 * H, acc, lane);
            acc = wmma_k256(h1full, Whh1L + wv * 16 * H, acc, lane);
            const int n = lane & 15, kh = lane >> 4;
#pragma unroll
            for (int j = 0; j < 8; ++j) gbuf[wv][j + 8 * kh][n] = acc[j];
        }
        __syncthreads();

        for (int e = tid; e < 256; e += NTHREADS) {
            int m = e >> 4, n = e & 15;
            float gv[4];
#pragma unroll
            for (int q = 0; q < 4; ++q)
                gv[q] = gbuf[q][m][n] + bias1L[q * 16 + n];
            float cn = fsigmoid(gv[1]) * cbuf1[m][n]
                     + fsigmoid(gv[0]) * ftanh(gv[2]);
            cbuf1[m][n] = cn;
            float hn = fsigmoid(gv[3]) * ftanh(cn);
            myH1[m * H + hbase + n] = (_Float16)hn;
        }
        group_barrier(cnt, gen, ++bgen);

        // gather full h1(t): needed as h1(t-1) next step and for the logits
        gather_to_lds(h1full, myH1, tid);
        __syncthreads();

        // distributed output projection: WG s handles batch row m = s.
        // Each lane sums a 16-wide k-chunk; 4-step xor-shuffle reduction.
        if (tid < 32) {
            const int m  = s;
            const int o  = lane >> 4;     // logit index 0/1
            const int kp = lane & 15;     // k-chunk
            float sum = 0.0f;
#pragma unroll
            for (int k = 0; k < 16; ++k)
                sum += (float)h1full[m * H + kp * 16 + k] * WoutL[o * H + kp * 16 + k];
#pragma unroll
            for (int mask = 8; mask >= 1; mask >>= 1)
                sum += __shfl_xor(sum, mask, 32);
            if (kp == 0)
                out[((g * BG + m) * T_LEN + t) * 2 + o] = sum + boutL[o];
        }
    }
}

extern "C" void kernel_launch(void* const* d_in, const int* in_sizes, int n_in,
                              void* d_out, int out_size, void* d_ws, size_t ws_size,
                              hipStream_t stream) {
    (void)in_sizes; (void)n_in; (void)out_size; (void)ws_size;
    const float* y     = (const float*)d_in[0];
    const float* W_ih0 = (const float*)d_in[1];
    const float* W_hh0 = (const float*)d_in[2];
    const float* b_ih0 = (const float*)d_in[3];
    const float* b_hh0 = (const float*)d_in[4];
    const float* W_ih1 = (const float*)d_in[5];
    const float* W_hh1 = (const float*)d_in[6];
    const float* b_ih1 = (const float*)d_in[7];
    const float* b_hh1 = (const float*)d_in[8];
    const float* W_out = (const float*)d_in[9];
    const float* b_out = (const float*)d_in[10];

    // zero the inter-workgroup barrier words (capture-safe graph memset node)
    (void)hipMemsetAsync(d_ws, 0, 256, stream);

    dim3 grid(GROUPS * WPG);   // 64 workgroups: 4 batch groups x 16 hidden slices
    dim3 block(NTHREADS);
    meta_lstm_kernel<<<grid, block, 0, stream>>>(
        y, W_ih0, W_hh0, b_ih0, b_hh0,
        W_ih1, W_hh1, b_ih1, b_hh1,
        W_out, b_out, (float*)d_out, d_ws);
}
